// PointNeXtSetAbstraction_26199300505655
// MI455X (gfx1250) — matrix operand order repaired and checked by reading, not compiled, not run on hardware
//
#include <hip/hip_runtime.h>
#include <hip/hip_bf16.h>

typedef __attribute__((ext_vector_type(16))) _Float16 v16h;
typedef __attribute__((ext_vector_type(8)))  float    v8f;

#define BB   8
#define NN   4096
#define CIN  64
#define MID  64
#define COUT 128
#define MM   1024      // N / STRIDE
#define KK   32        // NSAMPLE
#define NC1  (BB*MM*KK)   // 262144 columns for grouped MLP
#define NC2  (BB*MM)      // 8192 columns for skip GEMM
#define CP1  96           // 67 padded to multiple of 32
#define CP2  64

// ---------------- small utility kernels ----------------

__global__ void zero_kernel(float* p, int n) {
    int i = blockIdx.x * blockDim.x + threadIdx.x;
    if (i < n) p[i] = 0.0f;
}

__global__ void convert_pad_kernel(const float* __restrict__ src, _Float16* __restrict__ dst,
                                   int rows, int scols, int dcols) {
    int i = blockIdx.x * blockDim.x + threadIdx.x;
    if (i >= rows * dcols) return;
    int r = i / dcols, c = i % dcols;
    dst[i] = (c < scols) ? (_Float16)src[r * scols + c] : (_Float16)0.0f;
}

__global__ void finalize_bn_kernel(const float* __restrict__ sum, const float* __restrict__ sq,
                                   const float* __restrict__ g, const float* __restrict__ bta,
                                   float* __restrict__ scale, float* __restrict__ bias,
                                   int n, float invCount) {
    int i = blockIdx.x * blockDim.x + threadIdx.x;
    if (i >= n) return;
    float mean = sum[i] * invCount;
    float var  = sq[i] * invCount - mean * mean;
    float inv  = rsqrtf(var + 1e-5f);
    scale[i] = g[i] * inv;
    bias[i]  = bta[i] - mean * g[i] * inv;
}

// ---------------- farthest point sampling ----------------
// one block per batch; coords in LDS, per-thread min-dist in registers
__global__ void fps_kernel(const float* __restrict__ coords, int* __restrict__ sidx) {
    __shared__ float cx[NN], cy[NN], cz[NN];
    __shared__ float rv[256];
    __shared__ int   ri[256];
    __shared__ int   sFar;
    int t = threadIdx.x, b = blockIdx.x;
    const float* cb = coords + (size_t)b * NN * 3;

    float sx = 0.f, sy = 0.f, sz = 0.f;
#pragma unroll
    for (int j = 0; j < 16; ++j) {
        int i = t + 256 * j;
        float x = cb[i * 3 + 0], y = cb[i * 3 + 1], z = cb[i * 3 + 2];
        cx[i] = x; cy[i] = y; cz[i] = z;
        sx += x; sy += y; sz += z;
    }
    // mean reduction (3 components sequentially through rv)
    float mean[3]; float comp[3] = {sx, sy, sz};
    for (int cidx = 0; cidx < 3; ++cidx) {
        __syncthreads();
        rv[t] = comp[cidx];
        __syncthreads();
        for (int s2 = 128; s2 > 0; s2 >>= 1) {
            if (t < s2) rv[t] += rv[t + s2];
            __syncthreads();
        }
        mean[cidx] = rv[0] / (float)NN;
    }
    __syncthreads();

    float md[16];
    float bv = -1.0f; int bi = 0;
#pragma unroll
    for (int j = 0; j < 16; ++j) {
        int i = t + 256 * j;
        float dx = cx[i] - mean[0], dy = cy[i] - mean[1], dz = cz[i] - mean[2];
        float dd = dx * dx + dy * dy + dz * dz;
        md[j] = 3.4e38f;
        if (dd > bv) { bv = dd; bi = i; }
    }
    rv[t] = bv; ri[t] = bi;
    __syncthreads();
    for (int s2 = 128; s2 > 0; s2 >>= 1) {
        if (t < s2) {
            if (rv[t + s2] > rv[t] || (rv[t + s2] == rv[t] && ri[t + s2] < ri[t])) {
                rv[t] = rv[t + s2]; ri[t] = ri[t + s2];
            }
        }
        __syncthreads();
    }
    if (t == 0) sFar = ri[0];
    __syncthreads();

    for (int it = 0; it < MM; ++it) {
        int far = sFar;
        if (t == 0) sidx[b * MM + it] = far;
        float fx = cx[far], fy = cy[far], fz = cz[far];
        float lbv = -1.0f; int lbi = 0;
#pragma unroll
        for (int j = 0; j < 16; ++j) {
            int i = t + 256 * j;
            float dx = cx[i] - fx, dy = cy[i] - fy, dz = cz[i] - fz;
            float dd = dx * dx + dy * dy + dz * dz;
            md[j] = fminf(md[j], dd);
            if (md[j] > lbv) { lbv = md[j]; lbi = i; }
        }
        __syncthreads();
        rv[t] = lbv; ri[t] = lbi;
        __syncthreads();
        for (int s2 = 128; s2 > 0; s2 >>= 1) {
            if (t < s2) {
                if (rv[t + s2] > rv[t] || (rv[t + s2] == rv[t] && ri[t + s2] < ri[t])) {
                    rv[t] = rv[t + s2]; ri[t] = ri[t + s2];
                }
            }
            __syncthreads();
        }
        if (t == 0) sFar = ri[0];
        __syncthreads();
    }
}

__global__ void gather_sampled_kernel(const float* __restrict__ coords, const int* __restrict__ sidx,
                                      float* __restrict__ outSC) {
    int q = blockIdx.x * blockDim.x + threadIdx.x;   // < NC2
    if (q >= NC2) return;
    int b = q >> 10;
    int si = sidx[q];
    const float* cp = coords + ((size_t)b * NN + si) * 3;
    float* op = outSC + (size_t)q * 3;
    op[0] = cp[0]; op[1] = cp[1]; op[2] = cp[2];
}

__global__ void skip_gather_kernel(const float* __restrict__ feats, const int* __restrict__ sidx,
                                   _Float16* __restrict__ Sin) {
    int q = blockIdx.x * blockDim.x + threadIdx.x;   // < NC2
    if (q >= NC2) return;
    int b = q >> 10;
    int si = sidx[q];
    const float* fb = feats + (size_t)b * CIN * NN + si;
    _Float16* sp = Sin + (size_t)q * CP2;
#pragma unroll 4
    for (int c = 0; c < CIN; ++c) sp[c] = (_Float16)fb[(size_t)c * NN];
}

// ---------------- kNN (== ball query with knn fallback) ----------------
__global__ void knn_kernel(const float* __restrict__ coords, const float* __restrict__ sampled,
                           int* __restrict__ gidx) {
    __shared__ float hd[64 * KK];
    __shared__ int   hi[64 * KK];
    int t = threadIdx.x;
    int q = blockIdx.x * 64 + t;           // < NC2
    int b = q >> 10;
    float* d  = hd + t * KK;
    int*   id = hi + t * KK;
    const float* sp = sampled + (size_t)q * 3;
    float qx = sp[0], qy = sp[1], qz = sp[2];
    const float* cb = coords + (size_t)b * NN * 3;
    int cnt = 0;
    for (int n = 0; n < NN; ++n) {
        float dx = cb[n * 3 + 0] - qx;
        float dy = cb[n * 3 + 1] - qy;
        float dz = cb[n * 3 + 2] - qz;
        float d2 = dx * dx + dy * dy + dz * dz;
        if (cnt < KK) {
            int p = cnt++;
            while (p > 0 && d[p - 1] > d2) { d[p] = d[p - 1]; id[p] = id[p - 1]; --p; }
            d[p] = d2; id[p] = n;
        } else if (d2 < d[KK - 1]) {
            int p = KK - 1;
            while (p > 0 && d[p - 1] > d2) { d[p] = d[p - 1]; id[p] = id[p - 1]; --p; }
            d[p] = d2; id[p] = n;
        }
    }
    for (int k = 0; k < KK; ++k) gidx[(size_t)q * KK + k] = id[k];
}

// ---------------- build x = [rel(3) | feats(64) | pad] as f16, column-major ld=CP1 ----------------
__global__ void xbuild_kernel(const float* __restrict__ coords, const float* __restrict__ feats,
                              const int* __restrict__ gidx, const float* __restrict__ sampled,
                              _Float16* __restrict__ Xh) {
    int col = blockIdx.x * blockDim.x + threadIdx.x;   // < NC1
    if (col >= NC1) return;
    int group = col >> 5;
    int b = group >> 10;
    int gi = gidx[col];
    const float* cp = coords + ((size_t)b * NN + gi) * 3;
    const float* sp = sampled + (size_t)group * 3;
    _Float16* xp = Xh + (size_t)col * CP1;
    xp[0] = (_Float16)((cp[0] - sp[0]) * 5.0f);   // / RADIUS(0.2)
    xp[1] = (_Float16)((cp[1] - sp[1]) * 5.0f);
    xp[2] = (_Float16)((cp[2] - sp[2]) * 5.0f);
    const float* fb = feats + (size_t)b * CIN * NN + gi;
#pragma unroll 4
    for (int c = 0; c < CIN; ++c) xp[3 + c] = (_Float16)fb[(size_t)c * NN];
    for (int c = 3 + CIN; c < CP1; ++c) xp[c] = (_Float16)0.0f;
}

// ---------------- density weights: one wave per group of 32 ----------------
__global__ void density_kernel(const float* __restrict__ coords, const int* __restrict__ gidx,
                               float* __restrict__ wts) {
    __shared__ float ds[256 * KK];
    int lane = threadIdx.x & 31;
    int group = (blockIdx.x * blockDim.x + threadIdx.x) >> 5;   // < NC2
    int b = group >> 10;
    int gi = gidx[(size_t)group * KK + lane];
    const float* cp = coords + ((size_t)b * NN + gi) * 3;
    float x = cp[0], y = cp[1], z = cp[2];
    float* d = ds + (size_t)threadIdx.x * KK;
#pragma unroll
    for (int j = 0; j < KK; ++j) {
        float ox = __shfl(x, j, 32), oy = __shfl(y, j, 32), oz = __shfl(z, j, 32);
        float dx = x - ox, dy = y - oy, dz = z - oz;
        float dd = sqrtf(fmaxf(dx * dx + dy * dy + dz * dz, 1e-12f));
        d[j] = (j == lane) ? 3.0e38f : dd;
    }
    float kth = 0.0f;
    for (int it = 0; it < 16; ++it) {          // DENSITY_K = 16
        float mn = 3.0e38f; int mi = 0;
        for (int j = 0; j < KK; ++j) { float v = d[j]; if (v < mn) { mn = v; mi = j; } }
        kth = mn; d[mi] = 3.0e38f;
    }
    float tt = fmaxf(kth, 1e-8f);
    float raw = tt * tt * tt;                  // INTRINSIC_DIM = 3
    float s = raw;
#pragma unroll
    for (int off = 1; off < 32; off <<= 1) s += __shfl_xor(s, off, 32);
    wts[(size_t)group * KK + lane] = raw / fmaxf(s, 1e-8f);
}

// ---------------- WMMA lane mappings (contiguous k = 16*(lane/16) + e, same order as B) ----------------
__device__ __forceinline__ int kmapA(int lane, int e) {
    return ((lane >> 4) << 4) + e;
}

// ---------------- GEMM1: h_pre = W0(64x96) * X(96 x NC1), store f32 + stats ----------------
// One wave handles ALL 4 row-tiles of a column-tile: B loaded once per K-step.
__global__ void gemm1_kernel(const _Float16* __restrict__ W0h, const _Float16* __restrict__ Xh,
                             float* __restrict__ Hpre, float* __restrict__ sum1, float* __restrict__ sq1) {
    int ct = (blockIdx.x * blockDim.x + threadIdx.x) >> 5;   // column tile, exactly NC1/16
    int lane = threadIdx.x & 31;
    int n0 = ct << 4;
    int col = n0 + (lane & 15);
    int alow = lane & 15;
    v8f c[4];
#pragma unroll
    for (int rt = 0; rt < 4; ++rt) c[rt] = (v8f){};
#pragma unroll
    for (int ks = 0; ks < 3; ++ks) {
        int k0 = ks << 5;
        v16h bm;
        const _Float16* xp = Xh + (size_t)col * CP1 + k0 + ((lane >> 4) << 4);
#pragma unroll
        for (int e = 0; e < 16; ++e) bm[e] = xp[e];
#pragma unroll
        for (int rt = 0; rt < 4; ++rt) {
            int arow = (rt << 4) + alow;
            v16h a;
#pragma unroll
            for (int e = 0; e < 16; ++e) a[e] = W0h[arow * CP1 + k0 + kmapA(lane, e)];
            c[rt] = __builtin_amdgcn_wmma_f32_16x16x32_f16(false, a, false, bm, (short)0, c[rt], false, false);
        }
    }
#pragma unroll
    for (int rt = 0; rt < 4; ++rt) {
#pragma unroll
        for (int v = 0; v < 8; ++v) {
            int row = (rt << 4) + v + ((lane >> 4) << 3);
            Hpre[(size_t)col * MID + row] = c[rt][v];
        }
    }
#pragma unroll
    for (int rt = 0; rt < 4; ++rt) {
#pragma unroll
        for (int v = 0; v < 8; ++v) {
            float s = c[rt][v], q = c[rt][v] * c[rt][v];
#pragma unroll
            for (int off = 1; off < 16; off <<= 1) { s += __shfl_xor(s, off, 32); q += __shfl_xor(q, off, 32); }
            if ((lane & 15) == 0) {
                int row = (rt << 4) + v + ((lane >> 4) << 3);
                atomicAdd(&sum1[row], s);
                atomicAdd(&sq1[row], q);
            }
        }
    }
}

// ---------------- GEMM2: msg = W1(128x64) * relu(bn(h)); two-phase (stats / apply) ----------------
// One wave handles ALL 8 row-tiles: h_pre loaded + BN-converted once per K-step.
template <bool APPLY>
__global__ void gemm2_kernel(const _Float16* __restrict__ W1h, const float* __restrict__ Hpre,
                             const float* __restrict__ scale1, const float* __restrict__ bias1,
                             const float* __restrict__ scale2, const float* __restrict__ bias2,
                             const float* __restrict__ wts,
                             float* __restrict__ sum2, float* __restrict__ sq2,
                             float* __restrict__ Red) {
    int ct = (blockIdx.x * blockDim.x + threadIdx.x) >> 5;   // column tile, exactly NC1/16
    int lane = threadIdx.x & 31;
    int n0 = ct << 4;
    int col = n0 + (lane & 15);
    int alow = lane & 15;
    v8f c[8];
#pragma unroll
    for (int rt = 0; rt < 8; ++rt) c[rt] = (v8f){};
#pragma unroll
    for (int ks = 0; ks < 2; ++ks) {
        int k0 = ks << 5;
        int kb = k0 + ((lane >> 4) << 4);
        v16h bm;
        const float* hp = Hpre + (size_t)col * MID + kb;
#pragma unroll
        for (int e = 0; e < 16; ++e) {
            int ch = kb + e;
            bm[e] = (_Float16)fmaxf(hp[e] * scale1[ch] + bias1[ch], 0.0f);
        }
#pragma unroll
        for (int rt = 0; rt < 8; ++rt) {
            int arow = (rt << 4) + alow;
            v16h a;
#pragma unroll
            for (int e = 0; e < 16; ++e) a[e] = W1h[arow * MID + k0 + kmapA(lane, e)];
            c[rt] = __builtin_amdgcn_wmma_f32_16x16x32_f16(false, a, false, bm, (short)0, c[rt], false, false);
        }
    }
    if (!APPLY) {
#pragma unroll
        for (int rt = 0; rt < 8; ++rt) {
#pragma unroll
            for (int v = 0; v < 8; ++v) {
                float s = c[rt][v], q = c[rt][v] * c[rt][v];
#pragma unroll
                for (int off = 1; off < 16; off <<= 1) { s += __shfl_xor(s, off, 32); q += __shfl_xor(q, off, 32); }
                if ((lane & 15) == 0) {
                    int row = (rt << 4) + v + ((lane >> 4) << 3);
                    atomicAdd(&sum2[row], s);
                    atomicAdd(&sq2[row], q);
                }
            }
        }
    } else {
        int group = col >> 5;                  // all 16 cols of this tile share one group
        int bidx = group >> 10, midx = group & 1023;
        float w = wts[col];
#pragma unroll
        for (int rt = 0; rt < 8; ++rt) {
#pragma unroll
            for (int v = 0; v < 8; ++v) {
                int row = (rt << 4) + v + ((lane >> 4) << 3);
                float msg = fmaxf(scale2[row] * c[rt][v] + bias2[row], 0.0f) * w;
#pragma unroll
                for (int off = 1; off < 16; off <<= 1) msg += __shfl_xor(msg, off, 32);
                if ((lane & 15) == 0)
                    atomicAdd(&Red[(((size_t)bidx << 7) + row) * MM + midx], msg);
            }
        }
    }
}

// ---------------- skip GEMM: Ws(128x64) * skip_in(64 x NC2), store f32 + stats ----------------
__global__ void skip_gemm_kernel(const _Float16* __restrict__ Wsh, const _Float16* __restrict__ Sin,
                                 float* __restrict__ Spre, float* __restrict__ sums, float* __restrict__ sqs) {
    int ct = (blockIdx.x * blockDim.x + threadIdx.x) >> 5;   // column tile, exactly NC2/16
    int lane = threadIdx.x & 31;
    int n0 = ct << 4;
    int col = n0 + (lane & 15);
    int alow = lane & 15;
    v8f c[8];
#pragma unroll
    for (int rt = 0; rt < 8; ++rt) c[rt] = (v8f){};
#pragma unroll
    for (int ks = 0; ks < 2; ++ks) {
        int k0 = ks << 5;
        v16h bm;
        const _Float16* sp = Sin + (size_t)col * CP2 + k0 + ((lane >> 4) << 4);
#pragma unroll
        for (int e = 0; e < 16; ++e) bm[e] = sp[e];
#pragma unroll
        for (int rt = 0; rt < 8; ++rt) {
            int arow = (rt << 4) + alow;
            v16h a;
#pragma unroll
            for (int e = 0; e < 16; ++e) a[e] = Wsh[arow * CP2 + k0 + kmapA(lane, e)];
            c[rt] = __builtin_amdgcn_wmma_f32_16x16x32_f16(false, a, false, bm, (short)0, c[rt], false, false);
        }
    }
#pragma unroll
    for (int rt = 0; rt < 8; ++rt) {
#pragma unroll
        for (int v = 0; v < 8; ++v) {
            int row = (rt << 4) + v + ((lane >> 4) << 3);
            Spre[(size_t)col * COUT + row] = c[rt][v];
        }
    }
#pragma unroll
    for (int rt = 0; rt < 8; ++rt) {
#pragma unroll
        for (int v = 0; v < 8; ++v) {
            float s = c[rt][v], q = c[rt][v] * c[rt][v];
#pragma unroll
            for (int off = 1; off < 16; off <<= 1) { s += __shfl_xor(s, off, 32); q += __shfl_xor(q, off, 32); }
            if ((lane & 15) == 0) {
                int row = (rt << 4) + v + ((lane >> 4) << 3);
                atomicAdd(&sums[row], s);
                atomicAdd(&sqs[row], q);
            }
        }
    }
}

// ---------------- final: out = relu(reduced + bn(skip_pre)) ----------------
__global__ void combine_kernel(const float* __restrict__ Red, const float* __restrict__ Spre,
                               const float* __restrict__ scales, const float* __restrict__ biass,
                               float* __restrict__ out) {
    int idx = blockIdx.x * blockDim.x + threadIdx.x;   // < B*COUT*MM
    if (idx >= BB * COUT * MM) return;
    int b = idx >> 17;                 // / (COUT*MM)
    int o = (idx >> 10) & 127;
    int m = idx & 1023;
    int col = (b << 10) + m;
    float skip = scales[o] * Spre[(size_t)col * COUT + o] + biass[o];
    out[idx] = fmaxf(Red[idx] + skip, 0.0f);
}

// ---------------- host orchestration ----------------
extern "C" void kernel_launch(void* const* d_in, const int* in_sizes, int n_in,
                              void* d_out, int out_size, void* d_ws, size_t ws_size,
                              hipStream_t stream) {
    (void)in_sizes; (void)n_in; (void)out_size; (void)ws_size;
    const float* coords = (const float*)d_in[0];   // (B,N,3)
    const float* feats  = (const float*)d_in[1];   // (B,64,N)
    const float* W0     = (const float*)d_in[2];   // (64,67)
    const float* g0     = (const float*)d_in[3];
    const float* b0     = (const float*)d_in[4];
    const float* W1     = (const float*)d_in[5];   // (128,64)
    const float* g1     = (const float*)d_in[6];
    const float* b1     = (const float*)d_in[7];
    const float* Ws     = (const float*)d_in[8];   // (128,64)
    const float* gs     = (const float*)d_in[9];
    const float* bs     = (const float*)d_in[10];

    float* outSC  = (float*)d_out;                 // (B,M,3)  = 24576 floats
    float* outF   = (float*)d_out + BB * MM * 3;   // (B,128,M) = 1048576 floats

    char* w = (char*)d_ws;
    size_t off = 0;
    auto alloc = [&](size_t bytes) -> void* {
        void* p = w + off;
        off = (off + bytes + 255) & ~(size_t)255;
        return p;
    };
    int*      sidx  = (int*)alloc((size_t)NC2 * 4);
    int*      gidx  = (int*)alloc((size_t)NC1 * 4);
    float*    wts   = (float*)alloc((size_t)NC1 * 4);
    _Float16* Xh    = (_Float16*)alloc((size_t)CP1 * NC1 * 2);
    float*    Hpre  = (float*)alloc((size_t)MID * NC1 * 4);
    _Float16* Sin   = (_Float16*)alloc((size_t)CP2 * NC2 * 2);
    float*    Spre  = (float*)alloc((size_t)COUT * NC2 * 4);
    float*    Red   = (float*)alloc((size_t)BB * COUT * MM * 4);
    _Float16* W0h   = (_Float16*)alloc((size_t)MID * CP1 * 2);
    _Float16* W1h   = (_Float16*)alloc((size_t)COUT * MID * 2);
    _Float16* Wsh   = (_Float16*)alloc((size_t)COUT * CP2 * 2);
    float*    stats = (float*)alloc(2048 * 4);
    float* sum1 = stats + 0,   * sq1 = stats + 64;
    float* sum2 = stats + 128, * sq2 = stats + 256;
    float* sums = stats + 384, * sqs = stats + 512;
    float* scale1 = stats + 640,  * bias1 = stats + 704;
    float* scale2 = stats + 768,  * bias2 = stats + 896;
    float* scales = stats + 1024, * biass = stats + 1152;

    // zero atomic accumulators (must happen every call: graph replays)
    zero_kernel<<<4096, 256, 0, stream>>>(Red, BB * COUT * MM);
    zero_kernel<<<3, 256, 0, stream>>>(stats, 640);

    // weight conversion to f16 (padded)
    convert_pad_kernel<<<(MID * CP1 + 255) / 256, 256, 0, stream>>>(W0, W0h, MID, 67, CP1);
    convert_pad_kernel<<<(COUT * MID + 255) / 256, 256, 0, stream>>>(W1, W1h, COUT, MID, MID);
    convert_pad_kernel<<<(COUT * CP2 + 255) / 256, 256, 0, stream>>>(Ws, Wsh, COUT, CP2, CP2);

    // FPS -> sample indices
    fps_kernel<<<BB, 256, 0, stream>>>(coords, sidx);
    // gather sampled coords (first output) + skip features
    gather_sampled_kernel<<<NC2 / 256, 256, 0, stream>>>(coords, sidx, outSC);
    skip_gather_kernel<<<NC2 / 256, 256, 0, stream>>>(feats, sidx, Sin);
    // kNN grouping (ball query with knn fallback == exact top-32)
    knn_kernel<<<NC2 / 64, 64, 0, stream>>>(coords, outSC, gidx);
    // build grouped-input matrix and density weights
    xbuild_kernel<<<NC1 / 256, 256, 0, stream>>>(coords, feats, gidx, outSC, Xh);
    density_kernel<<<NC2 / 8, 256, 0, stream>>>(coords, gidx, wts);

    // GEMM1 (W0): one wave per 16-col tile covers all 4 row tiles; 8 waves per block
    gemm1_kernel<<<(NC1 / 16) / 8, 256, 0, stream>>>(W0h, Xh, Hpre, sum1, sq1);
    finalize_bn_kernel<<<1, 64, 0, stream>>>(sum1, sq1, g0, b0, scale1, bias1, MID, 1.0f / (float)NC1);

    // GEMM2 (W1): stats pass then apply pass (fused BN+ReLU+density-weighted K-reduction)
    gemm2_kernel<false><<<(NC1 / 16) / 8, 256, 0, stream>>>(W1h, Hpre, scale1, bias1,
                                                            scale2, bias2, wts, sum2, sq2, Red);
    finalize_bn_kernel<<<1, 128, 0, stream>>>(sum2, sq2, g1, b1, scale2, bias2, COUT, 1.0f / (float)NC1);
    gemm2_kernel<true><<<(NC1 / 16) / 8, 256, 0, stream>>>(W1h, Hpre, scale1, bias1,
                                                           scale2, bias2, wts, sum2, sq2, Red);

    // skip GEMM (Ws) + BN stats
    skip_gemm_kernel<<<(NC2 / 16) / 8, 256, 0, stream>>>(Wsh, Sin, Spre, sums, sqs);
    finalize_bn_kernel<<<1, 128, 0, stream>>>(sums, sqs, gs, bs, scales, biass, COUT, 1.0f / (float)NC2);

    // final combine
    combine_kernel<<<(BB * COUT * MM) / 256, 256, 0, stream>>>(Red, Spre, scales, biass, outF);
}
